// SimpleFocusedRNNCell_6339371729201
// MI455X (gfx1250) — matrix operand order repaired and checked by reading, not compiled, not run on hardware
//
#include <hip/hip_runtime.h>
#include <hip/hip_bf16.h>

typedef __attribute__((ext_vector_type(8)))  __bf16 v8bf;
typedef __attribute__((ext_vector_type(16))) __bf16 v16bf;
typedef __attribute__((ext_vector_type(8)))  float  v8f;
typedef __attribute__((ext_vector_type(4)))  float  v4f;

union V16 { v16bf v; v8bf h[2]; };

#define DIMN 1024      // D == U == 1024
#define BB   256
#define TT   512
#define NBLK 64        // persistent grid size (16 N-tiles x 4 M-tiles)

// ---------------------------------------------------------------------------
// CDNA5 async global->LDS copy (ASYNCcnt path) via inline asm.
// (The clang builtin exists but its pointer params use the unspellable
//  LangAS::cuda_device address space, so we emit the instruction directly.)
//   VDST = VGPR with LDS byte offset (low 32 bits of a generic LDS pointer)
//   VADDR = VGPR pair with 64-bit global address (GV mode)
// ---------------------------------------------------------------------------
#if defined(__HIP_DEVICE_COMPILE__)
__device__ __forceinline__ void g2l_b128(const void* g, void* l) {
    const unsigned lds = (unsigned)(unsigned long long)l;  // generic->LDS offset
    asm volatile("global_load_async_to_lds_b128 %0, %1, off"
                 :: "v"(lds), "v"(g)
                 : "memory");
}
__device__ __forceinline__ void g2l_wait() {
#if __has_builtin(__builtin_amdgcn_s_wait_asynccnt)
    __builtin_amdgcn_s_wait_asynccnt(0);
#else
    asm volatile("s_wait_asynccnt 0" ::: "memory");
#endif
}
#else
__device__ __forceinline__ void g2l_b128(const void* g, void* l) {
    *(v8bf*)l = *(const v8bf*)g;           // host-pass placeholder
}
__device__ __forceinline__ void g2l_wait() {}
#endif

// ---------------------------------------------------------------------------
// Kernel 1: build focus masks, apply to weights, store TRANSPOSED bf16 weights
//   y==0: WcT[u][k] = kernel[k][u] * mask_c(k,u)
//   y==1: WpT[u][k] = recurrent_kernel[k][u] * mask_p(k,u)
//   y==2: convert h0 -> bf16 ping-pong buffer #1; zero grid-barrier counter
// ---------------------------------------------------------------------------
__global__ __launch_bounds__(256) void prep_weights(
    const float* __restrict__ kernel, const float* __restrict__ rkernel,
    const float* __restrict__ mu_c, const float* __restrict__ sg_c,
    const float* __restrict__ mu_p, const float* __restrict__ sg_p,
    const float* __restrict__ h0,
    __bf16* __restrict__ WcT, __bf16* __restrict__ WpT,
    __bf16* __restrict__ hbf1, unsigned int* __restrict__ barrier_cnt)
{
    if (blockIdx.y == 2) {
        const int i = blockIdx.x * 256 + threadIdx.x;   // covers B*U exactly
        hbf1[i] = (__bf16)h0[i];
        if (i == 0) *barrier_cnt = 0u;
        return;
    }
    const int u = blockIdx.x;
    const bool prev = (blockIdx.y != 0);
    const float* W  = prev ? rkernel : kernel;
    const float  mu = prev ? mu_p[u] : mu_c[u];
    float s         = prev ? sg_p[u] : sg_c[u];
    __bf16* dst     = prev ? WpT : WcT;

    s = fminf(fmaxf(s, 0.01f), 1.0f);
    const float inv2s2 = 1.0f / (2.0f * s * s);
    const float step = 1.0f / (float)(DIMN - 1);

    float g[4];
    float ss = 0.0f;
    for (int j = 0; j < 4; ++j) {
        int i = threadIdx.x + 256 * j;
        float d = (float)i * step - mu;
        float e = __expf(-d * d * inv2s2);
        g[j] = e;
        ss += e * e;
    }
    __shared__ float red[256];
    red[threadIdx.x] = ss;
    __syncthreads();
    for (int off = 128; off > 0; off >>= 1) {
        if (threadIdx.x < off) red[threadIdx.x] += red[threadIdx.x + off];
        __syncthreads();
    }
    const float scale = 32.0f * rsqrtf(red[0]);   // sqrt(1024)/||g||
    for (int j = 0; j < 4; ++j) {
        int i = threadIdx.x + 256 * j;
        dst[(size_t)u * DIMN + i] = (__bf16)(W[(size_t)i * DIMN + u] * g[j] * scale);
    }
}

// ---------------------------------------------------------------------------
// Kernel 2: bulk GEMM  out = X @ Wc + bias  (hoisted non-recurrent term)
// Block: 256 threads = 8 waves; tile 128(M) x 128(N); wave tile 32x64.
// DOUBLE-BUFFERED LDS: tile k+1 staged (B via async global->LDS, A via
// fp32->bf16 VGPR path) while tile k feeds the WMMAs; one barrier per k-step.
// k-stride 40 -> conflict-free ds_load_b128 fragment reads.
// ---------------------------------------------------------------------------
#define LDT 40
#define NKT (DIMN / 32)

__global__ __launch_bounds__(256) void gemm_xw(
    const float* __restrict__ X,
    const __bf16* __restrict__ WT,
    const float* __restrict__ bias,
    float* __restrict__ Z)
{
    __shared__ __attribute__((aligned(16))) __bf16 la[2][128 * LDT];
    __shared__ __attribute__((aligned(16))) __bf16 lb[2][128 * LDT];

    const int tid  = threadIdx.x;
    const int lane = tid & 31;
    const int wave = tid >> 5;
    const int wm = wave >> 1;            // 0..3  -> M offset 32*wm
    const int wn = wave & 1;             // 0..1  -> N offset 64*wn
    const long m0 = (long)blockIdx.y * 128;
    const int  n0 = blockIdx.x * 128;

    v8f acc[2][4];
    for (int i = 0; i < 2; ++i)
        for (int j = 0; j < 4; ++j)
            for (int r = 0; r < 8; ++r) acc[i][j][r] = 0.0f;

    const int srow = tid >> 1;           // 0..127
    const int sseg = (tid & 1) * 16;     // 0 or 16
    const float*  asrc = X  + (m0 + srow) * (long)DIMN + sseg;
    const __bf16* bsrc = WT + (size_t)(n0 + srow) * DIMN + sseg;
    const int soff = srow * LDT + sseg;

    // ---- prologue: stage tile 0 ----
    g2l_b128(bsrc,     &lb[0][soff]);
    g2l_b128(bsrc + 8, &lb[0][soff + 8]);
    {
        v4f f0 = *(const v4f*)(asrc);
        v4f f1 = *(const v4f*)(asrc + 4);
        v4f f2 = *(const v4f*)(asrc + 8);
        v4f f3 = *(const v4f*)(asrc + 12);
        v8bf b0, b1;
        for (int r = 0; r < 4; ++r) {
            b0[r] = (__bf16)f0[r]; b0[4 + r] = (__bf16)f1[r];
            b1[r] = (__bf16)f2[r]; b1[4 + r] = (__bf16)f3[r];
        }
        *(v8bf*)(&la[0][soff])     = b0;
        *(v8bf*)(&la[0][soff + 8]) = b1;
    }
    g2l_wait();
    __syncthreads();

    const int arow  = 32 * wm + (lane & 15);
    const int akoff = (lane >> 4) * 8;       // A: K = akoff+{0..7}, akoff+{16..23}
    const int bkoff = (lane >> 4) * 8;       // B: K = bkoff+{0..7}, bkoff+{16..23}

    for (int kt = 0; kt < NKT; ++kt) {
        const int cur = kt & 1, nxt = cur ^ 1;
        const bool more = (kt + 1) < NKT;

        // ---- issue next tile's loads before compute ----
        v4f f0, f1, f2, f3;
        if (more) {
            const int knext = (kt + 1) * 32;
            g2l_b128(bsrc + knext,     &lb[nxt][soff]);      // async B -> LDS
            g2l_b128(bsrc + knext + 8, &lb[nxt][soff + 8]);
            const float* src = asrc + knext;
            f0 = *(const v4f*)(src);
            f1 = *(const v4f*)(src + 4);
            f2 = *(const v4f*)(src + 8);
            f3 = *(const v4f*)(src + 12);
            if (kt + 2 < NKT) __builtin_prefetch(src + 32, 0, 1);
        }

        // ---- compute on current tile ----
        V16 a[2], b[4];
        for (int mi = 0; mi < 2; ++mi) {
            const __bf16* p = &la[cur][(arow + 16 * mi) * LDT + akoff];
            a[mi].h[0] = *(const v8bf*)(p);
            a[mi].h[1] = *(const v8bf*)(p + 16);
        }
        for (int ni = 0; ni < 4; ++ni) {
            const __bf16* p = &lb[cur][(64 * wn + 16 * ni + (lane & 15)) * LDT + bkoff];
            b[ni].h[0] = *(const v8bf*)(p);
            b[ni].h[1] = *(const v8bf*)(p + 16);
        }
        for (int mi = 0; mi < 2; ++mi)
            for (int ni = 0; ni < 4; ++ni)
                acc[mi][ni] = __builtin_amdgcn_wmma_f32_16x16x32_bf16(
                    false, a[mi].v, false, b[ni].v, (short)0, acc[mi][ni], false, false);

        // ---- finish staging next tile ----
        if (more) {
            v8bf b0, b1;
            for (int r = 0; r < 4; ++r) {
                b0[r] = (__bf16)f0[r]; b0[4 + r] = (__bf16)f1[r];
                b1[r] = (__bf16)f2[r]; b1[4 + r] = (__bf16)f3[r];
            }
            *(v8bf*)(&la[nxt][soff])     = b0;
            *(v8bf*)(&la[nxt][soff + 8]) = b1;
            g2l_wait();
        }
        __syncthreads();
    }

    const int half = lane >> 4;
    const int nl   = lane & 15;
    for (int mi = 0; mi < 2; ++mi) {
        const long mbase = m0 + 32 * wm + 16 * mi + 8 * half;
        for (int ni = 0; ni < 4; ++ni) {
            const int n = n0 + 64 * wn + 16 * ni + nl;
            const float bv = bias[n];
            for (int r = 0; r < 8; ++r)
                Z[(mbase + r) * (long)DIMN + n] = acc[mi][ni][r] + bv;
        }
    }
}

// ---------------------------------------------------------------------------
// Kernel 3: PERSISTENT recurrence. One launch runs all 512 steps.
// 64 blocks x 128 threads. Block owns N-slice [n0, n0+64) of Wp, held in LDS
// (64 x 1024 bf16, k-stride 1032 -> conflict-free) for the whole run.
// h ping-pongs between two bf16 buffers; grid-wide barrier between steps.
//   out[b,t,n] = tanh(out[b,t,n] + sum_k h_{t-1}[b,k] * WpT[n,k])
// ---------------------------------------------------------------------------
#define LDK 1032   // padded k-stride for Wp slice in LDS

__device__ __forceinline__ void grid_barrier(unsigned int* cnt, unsigned int target) {
    __threadfence();                       // make hbf stores visible (release)
    __syncthreads();
    if (threadIdx.x == 0) {
        __hip_atomic_fetch_add(cnt, 1u, __ATOMIC_RELEASE, __HIP_MEMORY_SCOPE_AGENT);
        while (__hip_atomic_load(cnt, __ATOMIC_ACQUIRE, __HIP_MEMORY_SCOPE_AGENT) < target)
            __builtin_amdgcn_s_sleep(1);
    }
    __syncthreads();
    __threadfence();                       // acquire side
}

__global__ __launch_bounds__(128) void rnn_persistent(
    const __bf16* __restrict__ WpT,   // [U][U] bf16
    float* out,                       // [B,T,U] fp32, Z pre-filled
    __bf16* hbf,                      // 2 x [B][U] bf16 ping-pong
    unsigned int* barrier_cnt)
{
    __shared__ __attribute__((aligned(16))) __bf16 lb[64 * LDK];   // 129 KB

    const int tid  = threadIdx.x;
    const int lane = tid & 31;
    const int wave = tid >> 5;           // 0..3
    const int wm = wave >> 1, wn = wave & 1;
    const int n0 = blockIdx.x * 64;      // N slice owned by this block
    const int m0 = blockIdx.y * 64;      // M slice owned by this block
    const int rl   = lane & 15;
    const int half = lane >> 4;

    // ---- one-time preload of Wp slice into LDS (async DMA path) ----
    for (int c = tid; c < 64 * 128; c += 128) {
        const int row = c >> 7;               // 0..63
        const int kk  = (c & 127) * 8;        // 0..1016
        g2l_b128(&WpT[(size_t)(n0 + row) * DIMN + kk], &lb[row * LDK + kk]);
    }
    g2l_wait();
    __syncthreads();

    const size_t HS = (size_t)BB * DIMN;      // one h buffer
    for (int t = 0; t < TT; ++t) {
        const __bf16* hprev = hbf + ((t + 1) & 1) * HS;
        __bf16*       hnew  = hbf + (t & 1) * HS;
        float*        zt    = out + (size_t)t * DIMN;   // (b,n) at b*T*U + n

        v8f acc[2][2];
        for (int i = 0; i < 2; ++i)
            for (int j = 0; j < 2; ++j)
                for (int r = 0; r < 8; ++r) acc[i][j][r] = 0.0f;

        for (int k0 = 0; k0 < DIMN; k0 += 32) {
            V16 a[2], b[2];
            for (int mi = 0; mi < 2; ++mi) {
                const __bf16* p = hprev + (size_t)(m0 + 32 * wm + 16 * mi + rl) * DIMN
                                        + k0 + half * 8;
                a[mi].h[0] = *(const v8bf*)(p);
                a[mi].h[1] = *(const v8bf*)(p + 16);
            }
            for (int ni = 0; ni < 2; ++ni) {
                const __bf16* p = &lb[(32 * wn + 16 * ni + rl) * LDK + k0 + half * 8];
                b[ni].h[0] = *(const v8bf*)(p);
                b[ni].h[1] = *(const v8bf*)(p + 16);
            }
            for (int mi = 0; mi < 2; ++mi)
                for (int ni = 0; ni < 2; ++ni)
                    acc[mi][ni] = __builtin_amdgcn_wmma_f32_16x16x32_bf16(
                        false, a[mi].v, false, b[ni].v, (short)0, acc[mi][ni], false, false);
        }

        for (int mi = 0; mi < 2; ++mi) {
            const int mb = m0 + 32 * wm + 16 * mi + 8 * half;
            for (int ni = 0; ni < 2; ++ni) {
                const int n = n0 + 32 * wn + 16 * ni + rl;
                for (int r = 0; r < 8; ++r) {
                    const size_t zi = (size_t)(mb + r) * (TT * DIMN) + n;
                    const float hval = tanhf(acc[mi][ni][r] + zt[zi]);
                    zt[zi] = hval;                                    // output
                    hnew[(size_t)(mb + r) * DIMN + n] = (__bf16)hval; // recurrent state
                }
            }
        }
        grid_barrier(barrier_cnt, (unsigned int)(t + 1) * NBLK);
    }
}

// ---------------------------------------------------------------------------
extern "C" void kernel_launch(void* const* d_in, const int* in_sizes, int n_in,
                              void* d_out, int out_size, void* d_ws, size_t ws_size,
                              hipStream_t stream) {
    (void)in_sizes; (void)n_in; (void)out_size; (void)ws_size;
    const float* X     = (const float*)d_in[0];   // [B,T,D]
    const float* h0    = (const float*)d_in[1];   // [B,U]
    const float* kern  = (const float*)d_in[2];   // [D,U]
    const float* rkern = (const float*)d_in[3];   // [U,U]
    const float* bias  = (const float*)d_in[4];   // [U]
    const float* mu_c  = (const float*)d_in[5];
    const float* sg_c  = (const float*)d_in[6];
    const float* mu_p  = (const float*)d_in[7];
    const float* sg_p  = (const float*)d_in[8];
    float* out = (float*)d_out;                   // [B,T,U]

    char* ws = (char*)d_ws;
    __bf16* WcT = (__bf16*)(ws);                              // 2 MB
    __bf16* WpT = (__bf16*)(ws + (size_t)2 * 1024 * 1024);    // 2 MB
    __bf16* hbf = (__bf16*)(ws + (size_t)4 * 1024 * 1024);    // 2 x 512 KB
    unsigned int* cnt = (unsigned int*)(ws + (size_t)5 * 1024 * 1024 + 65536);

    // 1) masked + transposed bf16 weights, h0 -> bf16, reset barrier counter
    prep_weights<<<dim3(DIMN, 3), 256, 0, stream>>>(
        kern, rkern, mu_c, sg_c, mu_p, sg_p, h0,
        WcT, WpT, hbf + (size_t)BB * DIMN, cnt);
    // 2) hoisted bulk GEMM: out = X @ Wc + bias
    gemm_xw<<<dim3(DIMN / 128, (BB * TT) / 128), 256, 0, stream>>>(X, WcT, bias, out);
    // 3) single persistent kernel runs all 512 recurrent steps
    rnn_persistent<<<dim3(DIMN / 64, BB / 64), 128, 0, stream>>>(WpT, out, hbf, cnt);
}
